// DilatedRnnStack_53678501265422
// MI455X (gfx1250) — matrix operand order repaired and checked by reading, not compiled, not run on hardware
//
#include <hip/hip_runtime.h>
#include <hip/hip_bf16.h>
#include <stdint.h>

// ---------------------------------------------------------------------------
// Dilated LSTM stack on gfx1250: persistent bf16-WMMA kernel, one launch for
// all 64 recurrent (t, layer) phases. Split (arrive-early / wait-late) device
// barriers: the hidden-state K-chunks of each phase overlap the barrier wait.
// ---------------------------------------------------------------------------

typedef __attribute__((ext_vector_type(16))) __bf16        v16bf;
typedef __attribute__((ext_vector_type(8)))  float         v8f;
typedef __attribute__((ext_vector_type(4)))  float         v4f;
typedef __attribute__((ext_vector_type(4)))  unsigned int  v4u;
typedef __attribute__((ext_vector_type(2)))  unsigned int  v2u;

#define HSZ   256
#define SSZ   1024
#define OUTSZ 768
#define BSZ   64
#define TSZ   16
#define DIN   512
#define DOUT  512
#define KC    128   // K-chunk staged in LDS per double-buffer slot
#define NBLK  32    // persistent grid size (16 s-chunks x 2 batch halves)

union Frag { v4u u[2]; v16bf v; };

__device__ __forceinline__ unsigned short f2bf(float x) {
  uint32_t u = __float_as_uint(x);
  uint32_t r = u + 0x7FFFu + ((u >> 16) & 1u);   // round-to-nearest-even
  return (unsigned short)(r >> 16);
}

__device__ __forceinline__ float sigm(float x) { return 1.0f / (1.0f + __expf(-x)); }

// --------------------------- fp32 -> bf16 convert ---------------------------
__global__ void k_cvt(const float* __restrict__ src, unsigned short* __restrict__ dst, int n) {
  int i = blockIdx.x * blockDim.x + threadIdx.x;
  if (i < n) dst[i] = f2bf(src[i]);
}

__global__ void k_init(unsigned int* c) { *c = 0u; }

// ---- fused pack + LDS stage: one K-chunk of xh = [u | prevH | dH] in bf16 ---
// Each 128-wide chunk lies entirely inside one segment (128 | 512,768,256).
__device__ __forceinline__ void load_chunk_pack(
    unsigned short* __restrict__ dstLds,          // [32][KC] bf16
    const float* __restrict__ u, int ulen,        // [64][ulen] f32
    const float* __restrict__ hBufL,              // [T][64][256] f32
    int t, int d, int rb, int chOff, int tid) {
  for (int i = 0; i < 4; ++i) {
    int q = tid + i * 256;          // 1024 quads = 32 rows x 32 float4
    int row = q >> 5;               // 0..31
    int c = (q & 31) << 2;          // 0..124 step 4
    int col = chOff + c;
    v4f v;
    if (col < ulen) {
      v = *(const v4f*)&u[(size_t)(rb + row) * ulen + col];
    } else {
      int c2 = col - ulen;
      int hc = (c2 < HSZ) ? c2 : c2 - HSZ;
      int ts;
      if (c2 < HSZ) ts = (t > 0) ? (t - 1) : -1;                       // prevH
      else ts = ((t - d) >= 0) ? (t - d) : ((t > 0) ? (t - 1) : -1);   // dH
      if (ts >= 0) v = *(const v4f*)&hBufL[((size_t)ts * BSZ + rb + row) * HSZ + hc];
      else         v = (v4f){0.f, 0.f, 0.f, 0.f};
    }
    v2u p;
    p.x = (unsigned)f2bf(v.x) | ((unsigned)f2bf(v.y) << 16);
    p.y = (unsigned)f2bf(v.z) | ((unsigned)f2bf(v.w) << 16);
    *(v2u*)&dstLds[row * KC + c] = p;
  }
}

// --------------- persistent kernel: all 64 (t, layer) phases ----------------
// Block bid: s-chunk = bid&15 (64 gate columns), batch half = bid>>4 (32 rows).
// Wave w: mw = w&1 -> M-tile; nw = w>>1 -> s-tile; 4 gate accumulators/wave.
__global__ __launch_bounds__(256) void k_rnn(
    const float* __restrict__ x,                  // [T][64][512]
    const unsigned short* __restrict__ W0b, const unsigned short* __restrict__ W1b,
    const unsigned short* __restrict__ W2b, const unsigned short* __restrict__ W3b,
    const float* __restrict__ b0, const float* __restrict__ b1,
    const float* __restrict__ b2, const float* __restrict__ b3,
    float* __restrict__ hBuf,                     // [4][T][64][256]
    float* __restrict__ cBuf,                     // [4][T][64][1024]
    float* __restrict__ outB,                     // [4][64][768]
    unsigned short* __restrict__ bo,              // [T][64][768] bf16
    unsigned int* __restrict__ barCnt) {
  __shared__ __align__(16) unsigned short At[2][32 * KC];   // 2 x 8 KB

  const int tid  = threadIdx.x;
  const int wave = tid >> 5, lane = tid & 31;
  const int lm = lane & 15, lh = lane >> 4;
  const int mw = wave & 1;
  const int nw = wave >> 1;                    // 0..3
  const int bid = blockIdx.x;
  const int s0 = (bid & 15) * 64;
  const int rb = (bid >> 4) * 32;

  const v8f zero = {0.f, 0.f, 0.f, 0.f, 0.f, 0.f, 0.f, 0.f};
  unsigned int phase = 0;

  for (int t = 0; t < TSZ; ++t) {
#pragma unroll 1
    for (int l = 0; l < 4; ++l) {
      const int K    = (l == 0) ? (DIN + 2 * HSZ) : (OUTSZ + 2 * HSZ);
      const int ulen = (l == 0) ? DIN : OUTSZ;
      const int uch  = ulen / KC;              // u chunks: 4 or 6
      const int nch  = K / KC;                 // total chunks: 8 or 10
      const int d    = 1 << l;                 // dilations 1,2,4,8
      const float* u = (l == 0) ? (x + (size_t)t * BSZ * DIN)
                                : (outB + (size_t)(l - 1) * BSZ * OUTSZ);
      const unsigned short* W = (l == 0) ? W0b : (l == 1) ? W1b : (l == 2) ? W2b : W3b;
      const float* bias       = (l == 0) ? b0  : (l == 1) ? b1  : (l == 2) ? b2  : b3;
      float* cBufL = cBuf + (size_t)l * TSZ * BSZ * SSZ;
      float* hBufL = hBuf + (size_t)l * TSZ * BSZ * HSZ;
      float* outL  = outB + (size_t)l * BSZ * OUTSZ;

      const int scol = s0 + nw * 16 + lm;
      v8f acc[4];
      for (int g = 0; g < 4; ++g) acc[g] = zero;

      // ===== Part 1: hidden-state chunks [uch, nch) — data is >=4 phases old,
      // already ordered by earlier waits; overlaps this phase's barrier wait.
      load_chunk_pack(At[0], u, ulen, hBufL, t, d, rb, uch * KC, tid);
      __syncthreads();
      for (int ch = uch; ch < nch; ++ch) {
        const int cur = (ch - uch) & 1;
        if (ch + 1 < nch)
          load_chunk_pack(At[cur ^ 1], u, ulen, hBufL, t, d, rb, (ch + 1) * KC, tid);
        for (int ks = 0; ks < 4; ++ks) {
          Frag a;
          const unsigned short* ap = &At[cur][(mw * 16 + lm) * KC + ks * 32 + lh * 8];
          a.u[0] = *(const v4u*)ap;
          a.u[1] = *(const v4u*)(ap + 16);
          const int kg = ch * KC + ks * 32 + lh * 8;
          for (int g = 0; g < 4; ++g) {
            const unsigned short* bp = W + (size_t)(g * SSZ + scol) * K + kg;
            Frag bf;
            bf.u[0] = *(const v4u*)bp;
            bf.u[1] = *(const v4u*)(bp + 16);
            __builtin_prefetch(bp + KC, 0, 3);
            acc[g] = __builtin_amdgcn_wmma_f32_16x16x32_bf16(
                false, a.v, false, bf.v, (short)0, acc[g], false, false);
          }
        }
        __syncthreads();
      }

      // ===== Barrier WAIT (late): previous phase's epilogues must be visible
      // before reading outB[l-1] in the u chunks.
      if (phase > 0) {
        if (tid == 0) {
          while (__hip_atomic_load(barCnt, __ATOMIC_RELAXED,
                                   __HIP_MEMORY_SCOPE_AGENT) < phase * NBLK) {
            __builtin_amdgcn_s_sleep(1);
          }
        }
        __syncthreads();
        __threadfence();   // acquire on every wave (both CUs of the WGP)
      }

      // ===== Part 2: u chunks [0, uch) =====
      load_chunk_pack(At[0], u, ulen, hBufL, t, d, rb, 0, tid);
      __syncthreads();
      for (int ch = 0; ch < uch; ++ch) {
        const int cur = ch & 1;
        if (ch + 1 < uch)
          load_chunk_pack(At[cur ^ 1], u, ulen, hBufL, t, d, rb, (ch + 1) * KC, tid);
        for (int ks = 0; ks < 4; ++ks) {
          Frag a;
          const unsigned short* ap = &At[cur][(mw * 16 + lm) * KC + ks * 32 + lh * 8];
          a.u[0] = *(const v4u*)ap;
          a.u[1] = *(const v4u*)(ap + 16);
          const int kg = ch * KC + ks * 32 + lh * 8;
          for (int g = 0; g < 4; ++g) {
            const unsigned short* bp = W + (size_t)(g * SSZ + scol) * K + kg;
            Frag bf;
            bf.u[0] = *(const v4u*)bp;
            bf.u[1] = *(const v4u*)(bp + 16);
            __builtin_prefetch(bp + KC, 0, 3);
            acc[g] = __builtin_amdgcn_wmma_f32_16x16x32_bf16(
                false, a.v, false, bf.v, (short)0, acc[g], false, false);
          }
        }
        __syncthreads();
      }

      // --------------------- gate + state epilogue -------------------------
      const bool prev_ok = t > 0;
      const bool del_ok  = (t - d) >= 0;
      const float bb0 = bias[scol];
      const float bb1 = bias[SSZ + scol];
      const float bb2 = bias[2 * SSZ + scol];
      const float bb3 = bias[3 * SSZ + scol];
      for (int v = 0; v < 8; ++v) {
        const int b = rb + mw * 16 + v + lh * 8;       // batch row (C/D layout)
        float g0 = acc[0][v] + bb0;
        float g1 = acc[1][v] + bb1;
        float g2 = acc[2][v] + bb2;
        float g3 = acc[3][v] + bb3;
        float f  = sigm(g0 + 1.0f);
        float n  = tanhf(g1);
        float a  = sigm(g2);
        float o  = sigm(g3);
        float prevC = prev_ok ? cBufL[((size_t)(t - 1) * BSZ + b) * SSZ + scol] : 0.0f;
        float dC    = del_ok  ? cBufL[((size_t)(t - d) * BSZ + b) * SSZ + scol] : prevC;
        float wC    = del_ok  ? (a * prevC + (1.0f - a) * dC) : prevC;
        float cs    = prev_ok ? (f * wC + (1.0f - f) * n) : n;
        float whole = o * cs;
        cBufL[((size_t)t * BSZ + b) * SSZ + scol] = cs;
        if (scol < OUTSZ) {
          outL[(size_t)b * OUTSZ + scol] = whole;
          if (l == 3) {                      // fused inter-block residual
            float o1 = outB[(size_t)1 * BSZ * OUTSZ + (size_t)b * OUTSZ + scol];
            bo[((size_t)t * BSZ + b) * OUTSZ + scol] = f2bf(whole + o1);
          }
        } else {
          hBufL[((size_t)t * BSZ + b) * HSZ + (scol - OUTSZ)] = whole;
        }
      }

      // ===== Barrier ARRIVE (early): signal this phase done, don't wait. ====
      __threadfence();             // release epilogue writes (every wave)
      __syncthreads();
      if (tid == 0)
        __hip_atomic_fetch_add(barCnt, 1u, __ATOMIC_RELEASE, __HIP_MEMORY_SCOPE_AGENT);
      ++phase;
    }
  }
}

// ------------- adaptor: y[1024,512] = BO[1024,768] @ Wa.T + ba --------------
// grid = (16 row-chunks of 64, 4 col-chunks of 128), 256 threads.
__global__ __launch_bounds__(256) void k_adaptor(
    const unsigned short* __restrict__ A,    // [1024][768] bf16
    const unsigned short* __restrict__ Wa,   // [512][768]  bf16
    const float*          __restrict__ ba,   // [512]
    float* __restrict__ y) {                 // [1024][512]
  __shared__ __align__(16) unsigned short At[2][64 * KC];   // 2 x 16 KB

  const int tid  = threadIdx.x;
  const int wave = tid >> 5, lane = tid & 31;
  const int lm = lane & 15, lh = lane >> 4;
  const int mw = wave & 3, nw = wave >> 2;   // 4 M-tiles x 2 N-groups
  const int rowBase = blockIdx.x * 64;
  const int colBase = blockIdx.y * 128;
  const int K = OUTSZ;

  const v8f zero = {0.f, 0.f, 0.f, 0.f, 0.f, 0.f, 0.f, 0.f};
  v8f acc[4];
  for (int j = 0; j < 4; ++j) acc[j] = zero;

  for (int i = 0; i < 4; ++i) {
    int q = tid + i * 256;
    int row = q >> 4;
    int c = (q & 15) << 3;
    *(v4u*)&At[0][row * KC + c] = *(const v4u*)&A[(size_t)(rowBase + row) * K + c];
  }
  __syncthreads();

  const int nch = K / KC;                    // 6
  for (int ch = 0; ch < nch; ++ch) {
    const int cur = ch & 1;
    if (ch + 1 < nch) {
      const int nxt = cur ^ 1, kb = (ch + 1) * KC;
      for (int i = 0; i < 4; ++i) {
        int q = tid + i * 256;
        int row = q >> 4;
        int c = (q & 15) << 3;
        *(v4u*)&At[nxt][row * KC + c] =
            *(const v4u*)&A[(size_t)(rowBase + row) * K + kb + c];
      }
    }
    for (int ks = 0; ks < 4; ++ks) {
      Frag a;
      const unsigned short* ap = &At[cur][(mw * 16 + lm) * KC + ks * 32 + lh * 8];
      a.u[0] = *(const v4u*)ap;
      a.u[1] = *(const v4u*)(ap + 16);
      const int kg = ch * KC + ks * 32 + lh * 8;
      for (int j = 0; j < 4; ++j) {
        int col = colBase + (nw * 4 + j) * 16 + lm;
        const unsigned short* bp = Wa + (size_t)col * K + kg;
        Frag bf;
        bf.u[0] = *(const v4u*)bp;
        bf.u[1] = *(const v4u*)(bp + 16);
        acc[j] = __builtin_amdgcn_wmma_f32_16x16x32_bf16(
            false, a.v, false, bf.v, (short)0, acc[j], false, false);
      }
    }
    __syncthreads();
  }

  for (int j = 0; j < 4; ++j) {
    int col = colBase + (nw * 4 + j) * 16 + lm;
    float bb = ba[col];
    for (int v = 0; v < 8; ++v) {
      int row = rowBase + mw * 16 + v + lh * 8;
      y[(size_t)row * DOUT + col] = acc[j][v] + bb;
    }
  }
}

// ---------------------------------------------------------------------------
extern "C" void kernel_launch(void* const* d_in, const int* in_sizes, int n_in,
                              void* d_out, int out_size, void* d_ws, size_t ws_size,
                              hipStream_t stream) {
  (void)in_sizes; (void)n_in; (void)out_size; (void)ws_size;
  const float* x     = (const float*)d_in[0];
  const float* Wf[4] = {(const float*)d_in[1], (const float*)d_in[3],
                        (const float*)d_in[5], (const float*)d_in[7]};
  const float* bf[4] = {(const float*)d_in[2], (const float*)d_in[4],
                        (const float*)d_in[6], (const float*)d_in[8]};
  const float* Waf = (const float*)d_in[9];
  const float* ba  = (const float*)d_in[10];
  float* y = (float*)d_out;

  const int Kl[4] = {DIN + 2 * HSZ, OUTSZ + 2 * HSZ, OUTSZ + 2 * HSZ, OUTSZ + 2 * HSZ};

  // ---- workspace carve-up (256 B aligned) ----
  char* ws = (char*)d_ws;
  size_t off = 0;
  auto carve = [&](size_t bytes) -> char* {
    char* p = ws + off;
    off += (bytes + 255) & ~(size_t)255;
    return p;
  };
  unsigned short* Wb[4];
  for (int l = 0; l < 4; ++l) Wb[l] = (unsigned short*)carve((size_t)4 * SSZ * Kl[l] * 2);
  unsigned short* Wab = (unsigned short*)carve((size_t)DOUT * OUTSZ * 2);
  float* hBuf = (float*)carve((size_t)4 * TSZ * BSZ * HSZ * 4);
  float* cBuf = (float*)carve((size_t)4 * TSZ * BSZ * SSZ * 4);
  float* outB = (float*)carve((size_t)4 * BSZ * OUTSZ * 4);
  unsigned short* bo = (unsigned short*)carve((size_t)TSZ * BSZ * OUTSZ * 2);
  unsigned int* barCnt = (unsigned int*)carve(256);

  // ---- one-time weight conversion to bf16 (stays L2-resident thereafter) ----
  for (int l = 0; l < 4; ++l) {
    int n = 4 * SSZ * Kl[l];
    k_cvt<<<(n + 255) / 256, 256, 0, stream>>>(Wf[l], Wb[l], n);
  }
  {
    int n = DOUT * OUTSZ;
    k_cvt<<<(n + 255) / 256, 256, 0, stream>>>(Waf, Wab, n);
  }
  k_init<<<1, 1, 0, stream>>>(barCnt);

  // ---- single persistent launch for the whole recurrence ----
  k_rnn<<<NBLK, 256, 0, stream>>>(x,
                                  Wb[0], Wb[1], Wb[2], Wb[3],
                                  bf[0], bf[1], bf[2], bf[3],
                                  hBuf, cBuf, outB, bo, barCnt);

  // ---- batched adaptor GEMM over all timesteps ----
  k_adaptor<<<dim3(16, 4), 256, 0, stream>>>(bo, Wab, ba, y);
}
